// MultiHeadAttention_86741159510209
// MI455X (gfx1250) — matrix operand order repaired and checked by reading, not compiled, not run on hardware
//
#include <hip/hip_runtime.h>
#include <hip/hip_bf16.h>

typedef __bf16 bf16;
typedef __attribute__((ext_vector_type(16))) __bf16 v16bf;
typedef __attribute__((ext_vector_type(8)))  __bf16 v8bf;
typedef __attribute__((ext_vector_type(8)))  float  v8f;

#define D_MODEL 1024
#define N_HEADS 16
#define D_K     64
#define BATCH   4
#define SEQ     2048
#define M_TOT   (BATCH * SEQ)          // 8192

__device__ inline v8f zero8() {
    v8f z;
#pragma unroll
    for (int i = 0; i < 8; ++i) z[i] = 0.0f;
    return z;
}

__device__ inline v16bf join8(v8bf lo, v8bf hi) {
    return __builtin_shufflevector(lo, hi, 0,1,2,3,4,5,6,7,8,9,10,11,12,13,14,15);
}

// A-fragment (16x32 bf16): lane m = l&15, khalf = l>>4 holds
// K = khalf*16 + {0..15} (elems 0..15), contiguous, matching the B loads.
__device__ inline v16bf afrag_bf16(const bf16* rowptr, int k0, int lh) {
    const bf16* p = rowptr + k0 + lh * 16;
    v8bf lo = *(const v8bf*)(p);
    v8bf hi = *(const v8bf*)(p + 8);
    return join8(lo, hi);
}

// Same A-fragment built from fp32 memory, converted in-register to bf16.
__device__ inline v16bf afrag_f32(const float* rowptr, int k0, int lh) {
    const float4* p = (const float4*)(rowptr + k0 + lh * 16);
    float4 a = p[0], b = p[1], c = p[2], d = p[3];
    v16bf r;
    r[0]=(bf16)a.x; r[1]=(bf16)a.y; r[2] =(bf16)a.z; r[3] =(bf16)a.w;
    r[4]=(bf16)b.x; r[5]=(bf16)b.y; r[6] =(bf16)b.z; r[7] =(bf16)b.w;
    r[8]=(bf16)c.x; r[9]=(bf16)c.y; r[10]=(bf16)c.z; r[11]=(bf16)c.w;
    r[12]=(bf16)d.x;r[13]=(bf16)d.y;r[14]=(bf16)d.z; r[15]=(bf16)d.w;
    return r;
}

__device__ inline v8f wmma_bf16(v16bf a, v16bf b, v8f c) {
    return __builtin_amdgcn_wmma_f32_16x16x32_bf16(false, a, false, b, (short)0, c,
                                                   false, false);
}

// ---------------- fp32 -> bf16 weight conversion ----------------
__global__ __launch_bounds__(256) void cvt_kernel(const float* __restrict__ src,
                                                  bf16* __restrict__ dst, int n) {
    int i = (blockIdx.x * 256 + threadIdx.x) * 4;
    if (i + 3 < n) {
        float4 v = *(const float4*)(src + i);
        dst[i+0] = (bf16)v.x; dst[i+1] = (bf16)v.y;
        dst[i+2] = (bf16)v.z; dst[i+3] = (bf16)v.w;
    }
}

// ---------------- projection GEMM: C = X * W^T + bias ----------------
// Double-buffered (ping-pong) K loop: fragments for step k+32 are loaded
// before the WMMAs of step k execute, so loads overlap the matrix pipe.
// vmode==0: store bf16 [b,h,s,dk]; vmode==1: store bf16 transposed [b,h,dk,s].
__global__ __launch_bounds__(256) void qkv_proj_kernel(const float* __restrict__ X,
                                                       const bf16*  __restrict__ W,
                                                       const float* __restrict__ bias,
                                                       bf16* __restrict__ dst, int vmode) {
    int lane = threadIdx.x & 31, wv = threadIdx.x >> 5;
    int lm = lane & 15, lh = lane >> 4;
    int m0 = blockIdx.y * 128 + wv * 16;
    int n0 = blockIdx.x * 64;
    const float* xrow = X + (size_t)(m0 + lm) * D_MODEL;

    const bf16* wrow[4];
#pragma unroll
    for (int t = 0; t < 4; ++t)
        wrow[t] = W + (size_t)(n0 + t * 16 + lm) * D_MODEL + lh * 16;

    v8f acc[4];
#pragma unroll
    for (int t = 0; t < 4; ++t) acc[t] = zero8();

    v16bf aA = afrag_f32(xrow, 0, lh);
    v16bf bA[4];
#pragma unroll
    for (int t = 0; t < 4; ++t) bA[t] = *(const v16bf*)(wrow[t]);

    for (int k0 = 0; k0 < D_MODEL - 64; k0 += 64) {
        v16bf aB = afrag_f32(xrow, k0 + 32, lh);
        v16bf bB[4];
#pragma unroll
        for (int t = 0; t < 4; ++t) bB[t] = *(const v16bf*)(wrow[t] + k0 + 32);
#pragma unroll
        for (int t = 0; t < 4; ++t) acc[t] = wmma_bf16(aA, bA[t], acc[t]);

        aA = afrag_f32(xrow, k0 + 64, lh);
#pragma unroll
        for (int t = 0; t < 4; ++t) bA[t] = *(const v16bf*)(wrow[t] + k0 + 64);
#pragma unroll
        for (int t = 0; t < 4; ++t) acc[t] = wmma_bf16(aB, bB[t], acc[t]);
    }
    {   // final pair (k0 = D_MODEL-64): buffers A (loaded) then B (load now)
        v16bf aB = afrag_f32(xrow, D_MODEL - 32, lh);
        v16bf bB[4];
#pragma unroll
        for (int t = 0; t < 4; ++t) bB[t] = *(const v16bf*)(wrow[t] + D_MODEL - 32);
#pragma unroll
        for (int t = 0; t < 4; ++t) acc[t] = wmma_bf16(aA, bA[t], acc[t]);
#pragma unroll
        for (int t = 0; t < 4; ++t) acc[t] = wmma_bf16(aB, bB[t], acc[t]);
    }

#pragma unroll
    for (int t = 0; t < 4; ++t) {
        int n  = n0 + t * 16 + lm;
        int h  = n >> 6, dk = n & 63;
        float bs = bias[n];
#pragma unroll
        for (int r = 0; r < 8; ++r) {
            int m  = m0 + r + 8 * lh;
            int bb = m >> 11, s = m & (SEQ - 1);
            float v = acc[t][r] + bs;
            size_t idx;
            if (vmode) idx = (((size_t)bb * N_HEADS + h) * D_K + dk) * SEQ + s;
            else       idx = (((size_t)bb * N_HEADS + h) * SEQ + s) * D_K + dk;
            dst[idx] = (bf16)v;
        }
    }
}

// ---------------- flash attention (causal, online softmax) ----------------
// grid.x = SEQ/128 query blocks, grid.y = B*H. 8 waves, wave owns 16 q rows.
// V fragments are issued right after the S WMMAs and consumed only after the
// softmax VALU work + LDS transpose, hiding their latency completely.
__global__ __launch_bounds__(256) void flash_attn_kernel(const bf16* __restrict__ Qb,
                                                         const bf16* __restrict__ Kb,
                                                         const bf16* __restrict__ Vtb,
                                                         bf16* __restrict__ ctx) {
    __shared__ alignas(16) bf16 plds[8][16 * 64];   // per-wave P-transpose tile
    int lane = threadIdx.x & 31, wv = threadIdx.x >> 5;
    int lm = lane & 15, lh = lane >> 4;
    int bh = blockIdx.y;
    int q0 = blockIdx.x * 128 + wv * 16;
    const bf16* Qh = Qb  + (size_t)bh * SEQ * D_K;
    const bf16* Kh = Kb  + (size_t)bh * SEQ * D_K;
    const bf16* Vh = Vtb + (size_t)bh * D_K * SEQ;  // [dk][s]
    bf16* myP = &plds[wv][0];

    v16bf aq0 = afrag_bf16(Qh + (size_t)(q0 + lm) * D_K, 0,  lh);
    v16bf aq1 = afrag_bf16(Qh + (size_t)(q0 + lm) * D_K, 32, lh);

    v8f O[4];
#pragma unroll
    for (int t = 0; t < 4; ++t) O[t] = zero8();
    float mrow[8], lrow[8];
#pragma unroll
    for (int r = 0; r < 8; ++r) { mrow[r] = -3.0e38f; lrow[r] = 0.0f; }

    int nkb = (q0 + 16 + 63) >> 6;                  // causal: keys k <= q only
    for (int kb = 0; kb < nkb; ++kb) {
        int k0 = kb << 6;
        if (kb + 1 < nkb) {                         // prefetch next K/V blocks
            __builtin_prefetch(Kh + (size_t)(k0 + 64 + lm * 4) * D_K, 0, 1);
            __builtin_prefetch(Vh + (size_t)(lm * 4) * SEQ + k0 + 64, 0, 1);
        }

        // ---- load all K B-fragments for this block (one clause) ----
        v16bf kf[8];
#pragma unroll
        for (int t = 0; t < 4; ++t) {
            const bf16* kp = Kh + (size_t)(k0 + t * 16 + lm) * D_K + lh * 16;
            kf[2 * t]     = *(const v16bf*)(kp);
            kf[2 * t + 1] = *(const v16bf*)(kp + 32);
        }

        // ---- S = Q * K^T (16q x 64k), fp32 acc ----
        v8f sacc[4];
#pragma unroll
        for (int t = 0; t < 4; ++t) sacc[t] = zero8();
#pragma unroll
        for (int t = 0; t < 4; ++t) {
            sacc[t] = wmma_bf16(aq0, kf[2 * t],     sacc[t]);
            sacc[t] = wmma_bf16(aq1, kf[2 * t + 1], sacc[t]);
        }

        // ---- issue V B-fragments now; consumed after softmax ----
        v16bf vf[8];
#pragma unroll
        for (int t = 0; t < 4; ++t) {
            const bf16* vp = Vh + (size_t)(t * 16 + lm) * SEQ + k0 + lh * 16;
            vf[2 * t]     = *(const v16bf*)(vp);
            vf[2 * t + 1] = *(const v16bf*)(vp + 32);
        }

        // ---- scale + causal mask ----
#pragma unroll
        for (int t = 0; t < 4; ++t) {
            int k = k0 + t * 16 + lm;
#pragma unroll
            for (int r = 0; r < 8; ++r) {
                int q = q0 + r + 8 * lh;
                float s = sacc[t][r] * 0.125f;      // 1/sqrt(64)
                sacc[t][r] = (k > q) ? -3.0e38f : s;
            }
        }

        // ---- online softmax (rows live across the 16-lane halves) ----
        float bm[8];
#pragma unroll
        for (int r = 0; r < 8; ++r)
            bm[r] = fmaxf(fmaxf(sacc[0][r], sacc[1][r]), fmaxf(sacc[2][r], sacc[3][r]));
#pragma unroll
        for (int mk = 1; mk <= 8; mk <<= 1)
#pragma unroll
            for (int r = 0; r < 8; ++r) bm[r] = fmaxf(bm[r], __shfl_xor(bm[r], mk, 32));

        float corr[8];
#pragma unroll
        for (int r = 0; r < 8; ++r) {
            float mn = fmaxf(mrow[r], bm[r]);
            corr[r] = __expf(mrow[r] - mn);
            mrow[r] = mn;
        }
        float rs[8];
#pragma unroll
        for (int r = 0; r < 8; ++r) rs[r] = 0.0f;
#pragma unroll
        for (int t = 0; t < 4; ++t)
#pragma unroll
            for (int r = 0; r < 8; ++r) {
                float p = __expf(sacc[t][r] - mrow[r]);
                sacc[t][r] = p;
                rs[r] += p;
            }
#pragma unroll
        for (int mk = 1; mk <= 8; mk <<= 1)
#pragma unroll
            for (int r = 0; r < 8; ++r) rs[r] += __shfl_xor(rs[r], mk, 32);
#pragma unroll
        for (int r = 0; r < 8; ++r) lrow[r] = lrow[r] * corr[r] + rs[r];
#pragma unroll
        for (int t = 0; t < 4; ++t)
#pragma unroll
            for (int r = 0; r < 8; ++r) O[t][r] *= corr[r];

        // ---- P (C/D layout) -> LDS -> A-fragment layout ----
#pragma unroll
        for (int t = 0; t < 4; ++t)
#pragma unroll
            for (int r = 0; r < 8; ++r)
                myP[(r + 8 * lh) * 64 + t * 16 + lm] = (bf16)sacc[t][r];
        asm volatile("s_wait_dscnt 0" ::: "memory");
        v16bf ap0 = afrag_bf16(myP + lm * 64, 0,  lh);
        v16bf ap1 = afrag_bf16(myP + lm * 64, 32, lh);

        // ---- O += P * V (V fragments already in flight / arrived) ----
#pragma unroll
        for (int t = 0; t < 4; ++t) {
            O[t] = wmma_bf16(ap0, vf[2 * t],     O[t]);
            O[t] = wmma_bf16(ap1, vf[2 * t + 1], O[t]);
        }
    }

    // ---- normalize + store context [b, s, h*64+d] (bf16) ----
    int bb = bh >> 4, h = bh & 15;
#pragma unroll
    for (int t = 0; t < 4; ++t)
#pragma unroll
        for (int r = 0; r < 8; ++r) {
            int q = q0 + r + 8 * lh;
            int d = t * 16 + lm;
            float o = O[t][r] / lrow[r];
            ctx[((size_t)(bb * SEQ + q)) * D_MODEL + h * D_K + d] = (bf16)o;
        }
}

// ---------------- output GEMM: out = ctx * Wo^T + bo (fp32 out) ----------------
__global__ __launch_bounds__(256) void out_proj_kernel(const bf16* __restrict__ Xb,
                                                       const bf16* __restrict__ W,
                                                       const float* __restrict__ bias,
                                                       float* __restrict__ out) {
    int lane = threadIdx.x & 31, wv = threadIdx.x >> 5;
    int lm = lane & 15, lh = lane >> 4;
    int m0 = blockIdx.y * 128 + wv * 16;
    int n0 = blockIdx.x * 64;
    const bf16* xrow = Xb + (size_t)(m0 + lm) * D_MODEL;

    const bf16* wrow[4];
#pragma unroll
    for (int t = 0; t < 4; ++t)
        wrow[t] = W + (size_t)(n0 + t * 16 + lm) * D_MODEL + lh * 16;

    v8f acc[4];
#pragma unroll
    for (int t = 0; t < 4; ++t) acc[t] = zero8();

    v16bf aA = afrag_bf16(xrow, 0, lh);
    v16bf bA[4];
#pragma unroll
    for (int t = 0; t < 4; ++t) bA[t] = *(const v16bf*)(wrow[t]);

    for (int k0 = 0; k0 < D_MODEL - 64; k0 += 64) {
        v16bf aB = afrag_bf16(xrow, k0 + 32, lh);
        v16bf bB[4];
#pragma unroll
        for (int t = 0; t < 4; ++t) bB[t] = *(const v16bf*)(wrow[t] + k0 + 32);
#pragma unroll
        for (int t = 0; t < 4; ++t) acc[t] = wmma_bf16(aA, bA[t], acc[t]);

        aA = afrag_bf16(xrow, k0 + 64, lh);
#pragma unroll
        for (int t = 0; t < 4; ++t) bA[t] = *(const v16bf*)(wrow[t] + k0 + 64);
#pragma unroll
        for (int t = 0; t < 4; ++t) acc[t] = wmma_bf16(aB, bB[t], acc[t]);
    }
    {
        v16bf aB = afrag_bf16(xrow, D_MODEL - 32, lh);
        v16bf bB[4];
#pragma unroll
        for (int t = 0; t < 4; ++t) bB[t] = *(const v16bf*)(wrow[t] + D_MODEL - 32);
#pragma unroll
        for (int t = 0; t < 4; ++t) acc[t] = wmma_bf16(aA, bA[t], acc[t]);
#pragma unroll
        for (int t = 0; t < 4; ++t) acc[t] = wmma_bf16(aB, bB[t], acc[t]);
    }

#pragma unroll
    for (int t = 0; t < 4; ++t) {
        int n = n0 + t * 16 + lm;
        float bs = bias[n];
#pragma unroll
        for (int r = 0; r < 8; ++r) {
            int m = m0 + r + 8 * lh;
            out[(size_t)m * D_MODEL + n] = acc[t][r] + bs;
        }
    }
}

extern "C" void kernel_launch(void* const* d_in, const int* in_sizes, int n_in,
                              void* d_out, int out_size, void* d_ws, size_t ws_size,
                              hipStream_t stream) {
    const float* q  = (const float*)d_in[0];
    const float* k  = (const float*)d_in[1];
    const float* v  = (const float*)d_in[2];
    // d_in[3] = causal mask, recomputed analytically in-kernel
    const float* Wq = (const float*)d_in[4];
    const float* bq = (const float*)d_in[5];
    const float* Wk = (const float*)d_in[6];
    const float* bk = (const float*)d_in[7];
    const float* Wv = (const float*)d_in[8];
    const float* bv = (const float*)d_in[9];
    const float* Wo = (const float*)d_in[10];
    const float* bo = (const float*)d_in[11];

    bf16* ws = (bf16*)d_ws;
    const size_t WSZ = (size_t)D_MODEL * D_MODEL;          // 1M elems
    const size_t TSZ = (size_t)M_TOT * D_MODEL;            // 8.39M elems
    bf16* wq_bf = ws;               bf16* wk_bf = wq_bf + WSZ;
    bf16* wv_bf = wk_bf + WSZ;      bf16* wo_bf = wv_bf + WSZ;
    bf16* q_bf  = wo_bf + WSZ;      bf16* k_bf  = q_bf  + TSZ;
    bf16* vt_bf = k_bf  + TSZ;      bf16* ctx_bf= vt_bf + TSZ;

    int cvb = (int)(WSZ / (256 * 4));
    cvt_kernel<<<cvb, 256, 0, stream>>>(Wq, wq_bf, (int)WSZ);
    cvt_kernel<<<cvb, 256, 0, stream>>>(Wk, wk_bf, (int)WSZ);
    cvt_kernel<<<cvb, 256, 0, stream>>>(Wv, wv_bf, (int)WSZ);
    cvt_kernel<<<cvb, 256, 0, stream>>>(Wo, wo_bf, (int)WSZ);

    dim3 g(D_MODEL / 64, M_TOT / 128), blk(256);
    qkv_proj_kernel<<<g, blk, 0, stream>>>(q, wq_bf, bq, q_bf,  0);
    qkv_proj_kernel<<<g, blk, 0, stream>>>(k, wk_bf, bk, k_bf,  0);
    qkv_proj_kernel<<<g, blk, 0, stream>>>(v, wv_bf, bv, vt_bf, 1);

    dim3 ga(SEQ / 128, BATCH * N_HEADS);
    flash_attn_kernel<<<ga, blk, 0, stream>>>(q_bf, k_bf, vt_bf, ctx_bf);

    out_proj_kernel<<<g, blk, 0, stream>>>(ctx_bf, wo_bf, bo, (float*)d_out);
}
